// SNEA_88673894793753
// MI455X (gfx1250) — matrix operand order repaired and checked by reading, not compiled, not run on hardware
//
#include <hip/hip_runtime.h>
#include <hip/hip_bf16.h>

typedef __attribute__((ext_vector_type(16))) _Float16 v16h;
typedef __attribute__((ext_vector_type(8)))  float    v8f;

#define NNODES 100000
#define EPOS   800000
#define ENEG   200000
#define FDIM   64

// ---------------------------------------------------------------------------
// Elementwise: xm = x * sigmoid(feat_mask)
// ---------------------------------------------------------------------------
__global__ __launch_bounds__(256) void k_feat_mask(const float* __restrict__ x,
                                                   const float* __restrict__ fm,
                                                   float* __restrict__ xm, int total) {
    int i = blockIdx.x * 256 + threadIdx.x;
    if (i >= total) return;
    float s = 1.0f / (1.0f + expf(-fm[i & 63]));
    xm[i] = x[i] * s;
}

// ---------------------------------------------------------------------------
// Scatter-add mean aggregation (sum pass). 16 threads per edge, 4 feats each.
// mask == nullptr means all-ones.
// ---------------------------------------------------------------------------
__global__ __launch_bounds__(256) void k_scatter(const float* __restrict__ feat,
                                                 const int* __restrict__ ei,
                                                 const float* __restrict__ mask,
                                                 int E, float* __restrict__ sum,
                                                 float* __restrict__ cnt) {
    int t = blockIdx.x * 256 + threadIdx.x;
    int e = t >> 4;
    if (e >= E) return;
    int part = t & 15;
    float m = mask ? mask[e] : 1.0f;
    if (m == 0.0f) return;
    int s = ei[e];
    int d = ei[E + e];
    const float4 v = ((const float4*)(feat + (size_t)s * FDIM))[part];
    float* base = sum + (size_t)d * FDIM + part * 4;
    atomicAdd(base + 0, v.x);
    atomicAdd(base + 1, v.y);
    atomicAdd(base + 2, v.z);
    atomicAdd(base + 3, v.w);
    if (part == 0) atomicAdd(cnt + d, m);
}

__global__ __launch_bounds__(256) void k_divide(float* __restrict__ sum,
                                                const float* __restrict__ cnt, int total) {
    int i = blockIdx.x * 256 + threadIdx.x;
    if (i >= total) return;
    sum[i] = sum[i] / fmaxf(cnt[i >> 6], 1.0f);
}

// ---------------------------------------------------------------------------
// WMMA helpers: A-lane gather from padded LDS row (16x32 f16 A layout),
//               B-lane gather from row-major weight matrix (32x16 f16 B layout)
// ---------------------------------------------------------------------------
__device__ __forceinline__ v16h load_a_tile(const _Float16* row, int lane, int ks) {
    // row points at LDS row for M = lane%16 (padded stride applied by caller)
    v16h a;
    int kA = ks * 32 + ((lane < 16) ? 0 : 8);
#pragma unroll
    for (int v = 0; v < 4; ++v) {
        int k0 = kA + 2 * v;
        a[2 * v]     = row[k0];
        a[2 * v + 1] = row[k0 + 1];
        int k1 = kA + 16 + 2 * v;
        a[2 * (v + 4)]     = row[k1];
        a[2 * (v + 4) + 1] = row[k1 + 1];
    }
    return a;
}

__device__ __forceinline__ v16h load_b_tile(const float* __restrict__ wrow_base,
                                            int lane, int ks, int K) {
    // wrow_base = W + nrow*K ; B[k][n] = W[nrow][k]
    v16h b;
    int kB = ks * 32 + ((lane < 16) ? 0 : 8);
    const float* p = wrow_base + kB;
#pragma unroll
    for (int v = 0; v < 4; ++v) {
        b[2 * v]     = (_Float16)p[2 * v];
        b[2 * v + 1] = (_Float16)p[2 * v + 1];
        b[2 * (v + 4)]     = (_Float16)p[16 + 2 * v];
        b[2 * (v + 4) + 1] = (_Float16)p[16 + 2 * v + 1];
    }
    return b;
}

// ---------------------------------------------------------------------------
// conv_first GEMM: out[:, :32] = tanh([aggP | xm] @ Wp.T + bp)
//                  out[:, 32:] = tanh([aggN | xm] @ Wn.T + bn)
// K = 128. Block = 128 threads = 4 waves; block covers 16 nodes x 64 cols.
// ---------------------------------------------------------------------------
#define KP_FIRST 136
__global__ __launch_bounds__(128) void k_conv_first(const float* __restrict__ aggP,
                                                    const float* __restrict__ aggN,
                                                    const float* __restrict__ xm,
                                                    const float* __restrict__ Wp,
                                                    const float* __restrict__ bp,
                                                    const float* __restrict__ Wn,
                                                    const float* __restrict__ bn,
                                                    float* __restrict__ out,
                                                    float* __restrict__ out2) {
    __shared__ _Float16 As[2][16][KP_FIRST];
    int node0 = blockIdx.x * 16;
    int tid = threadIdx.x;
    for (int idx = tid; idx < 2 * 16 * 128; idx += 128) {
        int h = idx >> 11;
        int rem = idx & 2047;
        int m = rem >> 7;
        int k = rem & 127;
        size_t node = (size_t)(node0 + m);
        const float* agg = h ? aggN : aggP;
        float v = (k < 64) ? agg[node * FDIM + k] : xm[node * FDIM + (k - 64)];
        As[h][m][k] = (_Float16)v;
    }
    __syncthreads();

    int wave = tid >> 5;
    int lane = tid & 31;
    int half = wave >> 1;          // 0 = pos, 1 = neg
    int nOff = (wave & 1) * 16;    // weight-row block
    const float* W = half ? Wn : Wp;
    const float* B = half ? bn : bp;
    int mn = lane & 15;

    const _Float16* arow = &As[half][mn][0];
    const float* wrow = W + (size_t)(nOff + mn) * 128;

    v8f c = {};
#pragma unroll
    for (int ks = 0; ks < 4; ++ks) {
        v16h a = load_a_tile(arow, lane, ks);
        v16h b = load_b_tile(wrow, lane, ks, 128);
        c = __builtin_amdgcn_wmma_f32_16x16x32_f16(false, a, false, b, (short)0, c,
                                                   false, false);
    }

    int colOff = half * 32 + nOff;
    float bias = B[nOff + mn];
#pragma unroll
    for (int r = 0; r < 8; ++r) {
        int mm = r + ((lane >= 16) ? 8 : 0);
        float val = tanhf(c[r] + bias);
        size_t o = (size_t)(node0 + mm) * FDIM + colOff + mn;
        out[o] = val;
        if (out2) out2[o] = val;
    }
}

// ---------------------------------------------------------------------------
// conv_deep GEMM. AP = mean_agg(z, pos) (full 64), AN = mean_agg(z, neg).
// pos half: A = [AP[:,0:32] | AN[:,32:64] | z[:,0:32]]   (a_pp, a_nn, xp)
// neg half: A = [AP[:,32:64] | AN[:,0:32] | z[:,32:64]]  (a_pn, a_np, xn)
// K = 96. Optional tanh on output.
// ---------------------------------------------------------------------------
#define KP_DEEP 104
__global__ __launch_bounds__(128) void k_conv_deep(const float* __restrict__ AP,
                                                   const float* __restrict__ AN,
                                                   const float* __restrict__ z,
                                                   const float* __restrict__ Wp,
                                                   const float* __restrict__ bp,
                                                   const float* __restrict__ Wn,
                                                   const float* __restrict__ bn,
                                                   float* __restrict__ out,
                                                   int do_tanh) {
    __shared__ _Float16 As[2][16][KP_DEEP];
    int node0 = blockIdx.x * 16;
    int tid = threadIdx.x;
    for (int idx = tid; idx < 2 * 16 * 96; idx += 128) {
        int h = idx / 1536;
        int rem = idx - h * 1536;
        int m = rem / 96;
        int k = rem - m * 96;
        size_t node = (size_t)(node0 + m);
        float v;
        if (h == 0) {  // pos half
            if (k < 32)      v = AP[node * FDIM + k];
            else if (k < 64) v = AN[node * FDIM + k];        // a_nn lives in AN cols 32..63
            else             v = z[node * FDIM + (k - 64)];  // xp
        } else {       // neg half
            if (k < 32)      v = AP[node * FDIM + 32 + k];   // a_pn
            else if (k < 64) v = AN[node * FDIM + (k - 32)]; // a_np
            else             v = z[node * FDIM + (k - 32)];  // xn = z cols 32..63
        }
        As[h][m][k] = (_Float16)v;
    }
    __syncthreads();

    int wave = tid >> 5;
    int lane = tid & 31;
    int half = wave >> 1;
    int nOff = (wave & 1) * 16;
    const float* W = half ? Wn : Wp;
    const float* B = half ? bn : bp;
    int mn = lane & 15;

    const _Float16* arow = &As[half][mn][0];
    const float* wrow = W + (size_t)(nOff + mn) * 96;

    v8f c = {};
#pragma unroll
    for (int ks = 0; ks < 3; ++ks) {
        v16h a = load_a_tile(arow, lane, ks);
        v16h b = load_b_tile(wrow, lane, ks, 96);
        c = __builtin_amdgcn_wmma_f32_16x16x32_f16(false, a, false, b, (short)0, c,
                                                   false, false);
    }

    int colOff = half * 32 + nOff;
    float bias = B[nOff + mn];
#pragma unroll
    for (int r = 0; r < 8; ++r) {
        int mm = r + ((lane >= 16) ? 8 : 0);
        float val = c[r] + bias;
        if (do_tanh) val = tanhf(val);
        out[(size_t)(node0 + mm) * FDIM + colOff + mn] = val;
    }
}

// ---------------------------------------------------------------------------
// out = maybe_tanh(Zin) @ Ww.T + bw ; store or accumulate. K = 64, 64 cols.
// ---------------------------------------------------------------------------
#define KP_W 72
__global__ __launch_bounds__(128) void k_gemm_w(const float* __restrict__ Zin,
                                                const float* __restrict__ Ww,
                                                const float* __restrict__ bw,
                                                float* __restrict__ out,
                                                int tanh_in, int accum) {
    __shared__ _Float16 As[16][KP_W];
    int node0 = blockIdx.x * 16;
    int tid = threadIdx.x;
    for (int idx = tid; idx < 16 * 64; idx += 128) {
        int m = idx >> 6;
        int k = idx & 63;
        float v = Zin[(size_t)(node0 + m) * FDIM + k];
        if (tanh_in) v = tanhf(v);
        As[m][k] = (_Float16)v;
    }
    __syncthreads();

    int wave = tid >> 5;
    int lane = tid & 31;
    int nOff = wave * 16;
    int mn = lane & 15;

    const _Float16* arow = &As[mn][0];
    const float* wrow = Ww + (size_t)(nOff + mn) * 64;

    v8f c = {};
#pragma unroll
    for (int ks = 0; ks < 2; ++ks) {
        v16h a = load_a_tile(arow, lane, ks);
        v16h b = load_b_tile(wrow, lane, ks, 64);
        c = __builtin_amdgcn_wmma_f32_16x16x32_f16(false, a, false, b, (short)0, c,
                                                   false, false);
    }

    float bias = bw[nOff + mn];
#pragma unroll
    for (int r = 0; r < 8; ++r) {
        int mm = r + ((lane >= 16) ? 8 : 0);
        float val = c[r] + bias;
        size_t o = (size_t)(node0 + mm) * FDIM + nOff + mn;
        if (accum) out[o] += val;
        else       out[o] = val;
    }
}

// ---------------------------------------------------------------------------
// Edge sampling mask: mask[e] = sigmoid(dot(tmpz[src], tmpz[dst])) > rnd[e]
// ---------------------------------------------------------------------------
__global__ __launch_bounds__(256) void k_edge_mask(const float* __restrict__ tmpz,
                                                   const int* __restrict__ ei,
                                                   const float* __restrict__ rnd,
                                                   float* __restrict__ mask, int E) {
    int e = blockIdx.x * 256 + threadIdx.x;
    if (e >= E) return;
    int s = ei[e];
    int d = ei[E + e];
    const float4* zs = (const float4*)(tmpz + (size_t)s * FDIM);
    const float4* zd = (const float4*)(tmpz + (size_t)d * FDIM);
    float acc = 0.0f;
#pragma unroll
    for (int i = 0; i < 16; ++i) {
        float4 a = zs[i], b = zd[i];
        acc += a.x * b.x + a.y * b.y + a.z * b.z + a.w * b.w;
    }
    float sig = 1.0f / (1.0f + expf(-acc));
    mask[e] = (sig > rnd[e]) ? 1.0f : 0.0f;
}

// ---------------------------------------------------------------------------
// Host orchestration
// ---------------------------------------------------------------------------
static inline void agg_pass(const float* feat, const int* ei, const float* mask, int E,
                            float* sum, float* cnt, hipStream_t stream) {
    size_t NF = (size_t)NNODES * FDIM;
    hipMemsetAsync(sum, 0, NF * sizeof(float), stream);
    hipMemsetAsync(cnt, 0, (size_t)NNODES * sizeof(float), stream);
    int threads = E * 16;
    k_scatter<<<(threads + 255) / 256, 256, 0, stream>>>(feat, ei, mask, E, sum, cnt);
    k_divide<<<((int)NF + 255) / 256, 256, 0, stream>>>(sum, cnt, (int)NF);
}

extern "C" void kernel_launch(void* const* d_in, const int* in_sizes, int n_in,
                              void* d_out, int out_size, void* d_ws, size_t ws_size,
                              hipStream_t stream) {
    const float* x   = (const float*)d_in[0];
    const int*   pe  = (const int*)d_in[1];
    const int*   ne  = (const int*)d_in[2];
    const float* rp  = (const float*)d_in[3];
    const float* rn  = (const float*)d_in[4];
    const float* fm  = (const float*)d_in[5];
    const float* Wp0 = (const float*)d_in[6];
    const float* bp0 = (const float*)d_in[7];
    const float* Wn0 = (const float*)d_in[8];
    const float* bn0 = (const float*)d_in[9];
    const float* Wp1 = (const float*)d_in[10];
    const float* bp1 = (const float*)d_in[11];
    const float* Wn1 = (const float*)d_in[12];
    const float* bn1 = (const float*)d_in[13];
    const float* Ww  = (const float*)d_in[14];
    const float* bw  = (const float*)d_in[15];
    float* out = (float*)d_out;

    size_t NF = (size_t)NNODES * FDIM;
    float* ws   = (float*)d_ws;
    float* xm   = ws;
    float* bufA = ws + NF;
    float* bufB = ws + 2 * NF;
    float* bufZ = ws + 3 * NF;
    float* bufT = ws + 4 * NF;
    float* cnt  = ws + 5 * NF;
    float* pm   = cnt + NNODES;
    float* nm   = pm + EPOS;

    int nodeBlocks = NNODES / 16;  // 6250, exact

    // xm = x * sigmoid(feat_mask)
    k_feat_mask<<<((int)NF + 255) / 256, 256, 0, stream>>>(x, fm, xm, (int)NF);

    // ---- Stage 1: masks == ones ----
    agg_pass(xm, pe, nullptr, EPOS, bufA, cnt, stream);
    agg_pass(xm, ne, nullptr, ENEG, bufB, cnt, stream);
    k_conv_first<<<nodeBlocks, 128, 0, stream>>>(bufA, bufB, xm, Wp0, bp0, Wn0, bn0,
                                                 bufZ, nullptr);           // z
    agg_pass(bufZ, pe, nullptr, EPOS, bufA, cnt, stream);
    agg_pass(bufZ, ne, nullptr, ENEG, bufB, cnt, stream);
    k_conv_deep<<<nodeBlocks, 128, 0, stream>>>(bufA, bufB, bufZ, Wp1, bp1, Wn1, bn1,
                                                bufT, /*tanh=*/1);         // z2
    k_gemm_w<<<nodeBlocks, 128, 0, stream>>>(bufT, Ww, bw, bufA, /*tanh_in=*/0,
                                             /*accum=*/0);                 // tmp_z in bufA
    k_edge_mask<<<(EPOS + 255) / 256, 256, 0, stream>>>(bufA, pe, rp, pm, EPOS);
    k_edge_mask<<<(ENEG + 255) / 256, 256, 0, stream>>>(bufA, ne, rn, nm, ENEG);

    // ---- Stage 2: sampled masks ----
    agg_pass(xm, pe, pm, EPOS, bufA, cnt, stream);
    agg_pass(xm, ne, nm, ENEG, bufB, cnt, stream);
    k_conv_first<<<nodeBlocks, 128, 0, stream>>>(bufA, bufB, xm, Wp0, bp0, Wn0, bn0,
                                                 bufZ, out);               // h (+ xs copy)
    agg_pass(bufZ, pe, pm, EPOS, bufA, cnt, stream);
    agg_pass(bufZ, ne, nm, ENEG, bufB, cnt, stream);
    k_conv_deep<<<nodeBlocks, 128, 0, stream>>>(bufA, bufB, bufZ, Wp1, bp1, Wn1, bn1,
                                                bufT, /*tanh=*/0);         // h2 (raw)
    k_gemm_w<<<nodeBlocks, 128, 0, stream>>>(bufT, Ww, bw, out, /*tanh_in=*/1,
                                             /*accum=*/1);                 // xs += tanh(h2)@Ww.T + bw
}
